// Block_84585085927895
// MI455X (gfx1250) — matrix operand order repaired and checked by reading, not compiled, not run on hardware
//
#include <hip/hip_runtime.h>
#include <hip/hip_bf16.h>

// ---------------- problem constants ----------------
#define BATCH 4
#define TSEQ  2048
#define EMB   1024
#define HEADS 16
#define HD    64
#define FFD   4096
#define ROWS  (BATCH * TSEQ)          // 8192 token rows

// ---------------- WMMA types ----------------
typedef __attribute__((ext_vector_type(16))) __bf16 v16bf;
typedef __attribute__((ext_vector_type(8)))  float  v8f;
typedef __attribute__((ext_vector_type(4)))  int    v4i_;

union Frag16 {                 // one bf16 A/B WMMA operand (16x32 or 32x16), 32B
    uint4          q[2];
    unsigned short s[16];
    v16bf          v;
};
union AccF {                   // f32 16x16 C/D accumulator
    v8f   v;
    float f[8];
};

__device__ __forceinline__ unsigned short f2bf(float x) {
    unsigned int u = __float_as_uint(x);
    u += 0x7fffu + ((u >> 16) & 1u);   // round-to-nearest-even
    return (unsigned short)(u >> 16);
}

// ---------------- CDNA5 async global->LDS staging ----------------
#if defined(__AMDGCN__) &&                                            \
    __has_builtin(__builtin_amdgcn_global_load_async_to_lds_b128) &&  \
    __has_builtin(__builtin_amdgcn_s_wait_asynccnt)
#define USE_ASYNC_LDS 1
#else
#define USE_ASYNC_LDS 0
#endif

__device__ __forceinline__ void cp16_g2lds(unsigned short* dst_lds,
                                           const unsigned short* src_g) {
#if USE_ASYNC_LDS
    __builtin_amdgcn_global_load_async_to_lds_b128(
        (v4i_*)src_g, (v4i_*)dst_lds, 0, 0);
#else
    *(uint4*)dst_lds = *(const uint4*)src_g;
#endif
}

__device__ __forceinline__ void async_wait_tail6() {
#if USE_ASYNC_LDS
    __builtin_amdgcn_s_wait_asynccnt(6);   // newest tile may stay in flight
#endif
}
__device__ __forceinline__ void async_wait_all() {
#if USE_ASYNC_LDS
    __builtin_amdgcn_s_wait_asynccnt(0);
#endif
}

// ---------------- LayerNorm (f32 in -> bf16 out) ----------------
// one block (256 thr) per 1024-element row
__global__ void ln_bf16_kernel(const float* __restrict__ x,
                               const float* __restrict__ g,
                               const float* __restrict__ bta,
                               unsigned short* __restrict__ out) {
    const long row = blockIdx.x;
    const float* xr = x + row * (long)EMB;
    const int tid = threadIdx.x;

    float v[4];
    float s = 0.f, ss = 0.f;
#pragma unroll
    for (int i = 0; i < 4; ++i) {
        const int c = tid + i * 256;
        v[i] = xr[c];
        s  += v[i];
        ss += v[i] * v[i];
    }
#pragma unroll
    for (int m = 16; m >= 1; m >>= 1) {
        s  += __shfl_xor(s,  m);
        ss += __shfl_xor(ss, m);
    }
    __shared__ float bs[8], bss[8];
    const int wid = tid >> 5, lane = tid & 31;
    if (lane == 0) { bs[wid] = s; bss[wid] = ss; }
    __syncthreads();
    float ts = 0.f, tss = 0.f;
#pragma unroll
    for (int i = 0; i < 8; ++i) { ts += bs[i]; tss += bss[i]; }
    const float mu  = ts * (1.0f / EMB);
    const float var = tss * (1.0f / EMB) - mu * mu;
    const float rs  = rsqrtf(var + 1e-5f);
#pragma unroll
    for (int i = 0; i < 4; ++i) {
        const int c = tid + i * 256;
        out[row * (long)EMB + c] = f2bf((v[i] - mu) * rs * g[c] + bta[c]);
    }
}

// ---------------- f32 [K,N] -> bf16 [N,K] transpose (weights, once) --------
// block (32,8); grid (K/32, N/32, batch)
__global__ void cvt_t_bf16_kernel(const float* __restrict__ in,
                                  unsigned short* __restrict__ out,
                                  int K, int N) {
    __shared__ unsigned short tile[32][33];
    const long zoff = (long)blockIdx.z * K * N;
    in  += zoff;
    out += zoff;
    const int kb = blockIdx.x * 32;
    const int nb = blockIdx.y * 32;
#pragma unroll
    for (int i = threadIdx.y; i < 32; i += 8)
        tile[i][threadIdx.x] = f2bf(in[(long)(kb + i) * N + nb + threadIdx.x]);
    __syncthreads();
#pragma unroll
    for (int i = threadIdx.y; i < 32; i += 8)
        out[(long)(nb + i) * K + kb + threadIdx.x] = tile[threadIdx.x][i];
}

// ---------------- generic bf16 WMMA GEMM ----------------
// D[M,N] = act( A[M,K] * Bt[N,K]^T + bias + resid )
// block = 128 thr (4 waves); block tile 128x64 (wave tile 32x64); K step 32.
// Double-buffered LDS staging via async global->LDS b128 when available.
template <bool OUT_BF16, bool RELU, bool RESID>
__global__ void gemm_wmma_kernel(const unsigned short* __restrict__ A,
                                 const unsigned short* __restrict__ Bt,
                                 const float* __restrict__ bias,
                                 const float* __restrict__ resid,
                                 void* __restrict__ outv,
                                 int M, int N, int K,
                                 long strideB, long strideC) {
    constexpr int LDS_STR = 40;                     // ushort stride, 16B-aligned
    __shared__ unsigned short As[2][128 * LDS_STR]; // A tile  [row][k]
    __shared__ unsigned short Bs[2][64 * LDS_STR];  // Bt tile [n][k]

    const int tid  = threadIdx.x;
    const int lane = tid & 31;
    const int w    = tid >> 5;
    const int m0   = blockIdx.y * 128;
    const int n0   = blockIdx.x * 64;
    const unsigned short* Bp = Bt + (long)blockIdx.z * strideB;

    AccF acc[2][4];
#pragma unroll
    for (int ms = 0; ms < 2; ++ms)
#pragma unroll
        for (int nt = 0; nt < 4; ++nt)
#pragma unroll
            for (int r = 0; r < 8; ++r) acc[ms][nt].f[r] = 0.f;

    const int ac0 = (lane >> 4) * 8;       // A frag elems 0-7 : K=ac0..ac0+7
    const int bc0 = (lane >> 4) * 8;       // B frag elems 0-7 : K=bc0..bc0+7, 8-15: K=bc0+16..bc0+23

    // stage one 128x32 A tile + 64x32 B tile into buffer `buf` (6 ops/thread)
    auto stage = [&](int buf, int k0) {
        unsigned short* as = As[buf];
        unsigned short* bs2 = Bs[buf];
        for (int c = tid; c < 512; c += 128) {           // A: 512 16B chunks
            const int r  = c >> 2;
            const int kc = (c & 3) * 8;
            cp16_g2lds(&as[r * LDS_STR + kc],
                       A + (long)(m0 + r) * K + k0 + kc);
        }
        for (int c = tid; c < 256; c += 128) {           // B: 256 16B chunks
            const int n  = c >> 2;
            const int kc = (c & 3) * 8;
            cp16_g2lds(&bs2[n * LDS_STR + kc],
                       Bp + (long)(n0 + n) * K + k0 + kc);
        }
    };

    stage(0, 0);
    int it = 0;
    for (int k0 = 0; k0 < K; k0 += 32, ++it) {
        const int  cur = it & 1;
        const bool pre = (k0 + 32) < K;
        if (pre) stage(cur ^ 1, k0 + 32);
        if (pre) async_wait_tail6(); else async_wait_all();
        __syncthreads();

        const unsigned short* as = As[cur];
        const unsigned short* bs2 = Bs[cur];
        Frag16 af[2];
#pragma unroll
        for (int ms = 0; ms < 2; ++ms) {
            const int arow = w * 32 + ms * 16 + (lane & 15);
            af[ms].q[0] = *(const uint4*)(&as[arow * LDS_STR + ac0]);
            af[ms].q[1] = *(const uint4*)(&as[arow * LDS_STR + ac0 + 16]);
        }
#pragma unroll
        for (int nt = 0; nt < 4; ++nt) {
            Frag16 bf_;
            const int brow = nt * 16 + (lane & 15);
            bf_.q[0] = *(const uint4*)(&bs2[brow * LDS_STR + bc0]);
            bf_.q[1] = *(const uint4*)(&bs2[brow * LDS_STR + bc0 + 16]);
#pragma unroll
            for (int ms = 0; ms < 2; ++ms)
                acc[ms][nt].v = __builtin_amdgcn_wmma_f32_16x16x32_bf16(
                    false, af[ms].v, false, bf_.v, (short)0, acc[ms][nt].v,
                    false, false);
        }
        __syncthreads();
    }

    // epilogue
#pragma unroll
    for (int ms = 0; ms < 2; ++ms)
#pragma unroll
        for (int nt = 0; nt < 4; ++nt)
#pragma unroll
            for (int r = 0; r < 8; ++r) {
                const int grow = m0 + w * 32 + ms * 16 + r + ((lane >> 4) << 3);
                const int col  = n0 + nt * 16 + (lane & 15);
                float val = acc[ms][nt].f[r];
                if (bias)  val += bias[col];
                if (RELU)  val = fmaxf(val, 0.f);
                if (RESID) val += resid[(long)grow * N + col];
                if (OUT_BF16) {
                    unsigned short* out =
                        (unsigned short*)outv + (long)blockIdx.z * strideC;
                    out[(long)grow * N + col] = f2bf(val);
                } else {
                    float* out = (float*)outv + (long)blockIdx.z * strideC;
                    out[(long)grow * N + col] = val;
                }
            }
}

// ---------------- flash attention (causal, HD=64) ----------------
// q/k/v layout: [H, B, T, HD] bf16. out: [B, T, EMB] bf16 (heads concat).
// block = 128 thr; wave w owns query tile q0 = blockIdx.x*64 + w*16.
__global__ void attn_wmma_kernel(const unsigned short* __restrict__ qb,
                                 const unsigned short* __restrict__ kb,
                                 const unsigned short* __restrict__ vb,
                                 unsigned short* __restrict__ outb) {
    const int lane = threadIdx.x & 31;
    const int w    = threadIdx.x >> 5;
    const int bh   = blockIdx.y;
    const int bi   = bh / HEADS;
    const int hi   = bh % HEADS;
    const int q0   = blockIdx.x * 64 + w * 16;

    const long base = ((long)(hi * BATCH + bi) * TSEQ) * HD;
    const unsigned short* qp = qb + base;
    const unsigned short* kp = kb + base;
    const unsigned short* vp = vb + base;

    __shared__ unsigned short Ps[4][16 * 32];   // per-wave P tile (16x32)
    unsigned short* myP = Ps[w];

    const int half8 = (lane >> 4) * 8;
    const int kc    = (lane >> 4) * 8;

    // Q A-fragments: d = 0..31 and 32..63
    Frag16 qf[2];
    {
        const long trow = q0 + (lane & 15);
#pragma unroll
        for (int f = 0; f < 2; ++f) {
            qf[f].q[0] = *(const uint4*)(qp + trow * HD + f * 32 + half8);
            qf[f].q[1] = *(const uint4*)(qp + trow * HD + f * 32 + 16 + half8);
        }
    }

    float mrow[8], lrow[8];
    AccF  o[4];
#pragma unroll
    for (int r = 0; r < 8; ++r) { mrow[r] = -1e30f; lrow[r] = 0.f; }
#pragma unroll
    for (int nt = 0; nt < 4; ++nt)
#pragma unroll
        for (int r = 0; r < 8; ++r) o[nt].f[r] = 0.f;

    const int sEnd = q0 + 16;                 // need keys s <= q0+15
    for (int s0 = 0; s0 < sEnd; s0 += 32) {
        // prefetch next key/value block into cache (global_prefetch_b8)
        if (s0 + 32 < sEnd) {
            __builtin_prefetch(kp + (long)(s0 + 32 + lane) * HD, 0, 0);
            __builtin_prefetch(vp + (long)(s0 + 32 + lane) * HD, 0, 0);
        }

        // ---- S = Q * K^T for 16x32 key block (two 16x16 n-tiles) ----
        AccF s[2];
#pragma unroll
        for (int j = 0; j < 2; ++j) {
            const long srow = s0 + j * 16 + (lane & 15);
            Frag16 kf0, kf1;                  // B-frags for d 0..31 / 32..63
            kf0.q[0] = *(const uint4*)(kp + srow * HD + kc);
            kf0.q[1] = *(const uint4*)(kp + srow * HD + kc + 16);
            kf1.q[0] = *(const uint4*)(kp + srow * HD + 32 + kc);
            kf1.q[1] = *(const uint4*)(kp + srow * HD + 32 + kc + 16);
            v8f sa;
            AccF z;
#pragma unroll
            for (int r = 0; r < 8; ++r) z.f[r] = 0.f;
            sa = __builtin_amdgcn_wmma_f32_16x16x32_bf16(
                false, qf[0].v, false, kf0.v, (short)0, z.v, false, false);
            sa = __builtin_amdgcn_wmma_f32_16x16x32_bf16(
                false, qf[1].v, false, kf1.v, (short)0, sa, false, false);
            s[j].v = sa;
        }

        // ---- scale + causal mask ----
        const bool needMask = (s0 + 31) > q0;
#pragma unroll
        for (int j = 0; j < 2; ++j)
#pragma unroll
            for (int r = 0; r < 8; ++r) {
                float val = s[j].f[r] * 0.125f;     // 1/sqrt(64)
                if (needMask) {
                    const int scol = s0 + j * 16 + (lane & 15);
                    const int trow = q0 + r + ((lane >> 4) << 3);
                    if (scol > trow) val = -1e30f;
                }
                s[j].f[r] = val;
            }

        // ---- online softmax (rows live on 16-lane halves) ----
        float alpha[8];
#pragma unroll
        for (int r = 0; r < 8; ++r) {
            float pm = fmaxf(s[0].f[r], s[1].f[r]);
#pragma unroll
            for (int m = 1; m < 16; m <<= 1) pm = fmaxf(pm, __shfl_xor(pm, m));
            const float mn = fmaxf(mrow[r], pm);
            alpha[r] = __expf(mrow[r] - mn);
            mrow[r]  = mn;
            const float p0 = __expf(s[0].f[r] - mn);
            const float p1 = __expf(s[1].f[r] - mn);
            s[0].f[r] = p0;
            s[1].f[r] = p1;
            float ps = p0 + p1;
#pragma unroll
            for (int m = 1; m < 16; m <<= 1) ps += __shfl_xor(ps, m);
            lrow[r] = lrow[r] * alpha[r] + ps;
        }

        // ---- P: D-layout -> A-layout via LDS (transpose across lanes) ----
#pragma unroll
        for (int j = 0; j < 2; ++j)
#pragma unroll
            for (int r = 0; r < 8; ++r) {
                const int mR = r + ((lane >> 4) << 3);
                myP[mR * 32 + j * 16 + (lane & 15)] = f2bf(s[j].f[r]);
            }

        // rescale running output
#pragma unroll
        for (int nt = 0; nt < 4; ++nt)
#pragma unroll
            for (int r = 0; r < 8; ++r) o[nt].f[r] *= alpha[r];

        Frag16 pf;
        {
            const int prow = lane & 15;
            pf.q[0] = *(const uint4*)(&myP[prow * 32 + half8]);
            pf.q[1] = *(const uint4*)(&myP[prow * 32 + 16 + half8]);
        }

        // ---- O += P * V (V block 32x64, 4 n-tiles) ----
#pragma unroll
        for (int nt = 0; nt < 4; ++nt) {
            Frag16 vf;
            const int d  = nt * 16 + (lane & 15);
            const int sb = s0 + half8;
#pragma unroll
            for (int e = 0; e < 16; ++e)
                vf.s[e] = vp[(long)(sb + e + (e & 8)) * HD + d];
            o[nt].v = __builtin_amdgcn_wmma_f32_16x16x32_bf16(
                false, pf.v, false, vf.v, (short)0, o[nt].v, false, false);
        }
    }

    // ---- normalize + store (heads concatenated: [B,T,EMB] bf16) ----
#pragma unroll
    for (int nt = 0; nt < 4; ++nt)
#pragma unroll
        for (int r = 0; r < 8; ++r) {
            const int t = q0 + r + ((lane >> 4) << 3);
            const int d = nt * 16 + (lane & 15);
            const float val = o[nt].f[r] / lrow[r];
            outb[((long)(bi * TSEQ + t)) * EMB + hi * HD + d] = f2bf(val);
        }
}

// ---------------- launcher ----------------
extern "C" void kernel_launch(void* const* d_in, const int* in_sizes, int n_in,
                              void* d_out, int out_size, void* d_ws, size_t ws_size,
                              hipStream_t stream) {
    (void)in_sizes; (void)n_in; (void)out_size; (void)ws_size;
    const float* x   = (const float*)d_in[0];
    const float* Wq  = (const float*)d_in[1];
    const float* Wk  = (const float*)d_in[2];
    const float* Wv  = (const float*)d_in[3];
    const float* Wo  = (const float*)d_in[4];
    const float* bo  = (const float*)d_in[5];
    const float* W1  = (const float*)d_in[6];
    const float* b1  = (const float*)d_in[7];
    const float* W2  = (const float*)d_in[8];
    const float* b2  = (const float*)d_in[9];
    const float* g1  = (const float*)d_in[10];
    const float* be1 = (const float*)d_in[11];
    const float* g2  = (const float*)d_in[12];
    const float* be2 = (const float*)d_in[13];

    char* ws = (char*)d_ws;
    size_t off = 0;
    auto take = [&](size_t bytes) -> char* {
        char* p = ws + off;
        off = (off + bytes + 255) & ~(size_t)255;
        return p;
    };
    unsigned short* hbf   = (unsigned short*)take((size_t)ROWS * EMB * 2);
    unsigned short* wqt   = (unsigned short*)take((size_t)HEADS * EMB * HD * 2);
    unsigned short* wkt   = (unsigned short*)take((size_t)HEADS * EMB * HD * 2);
    unsigned short* wvt   = (unsigned short*)take((size_t)HEADS * EMB * HD * 2);
    unsigned short* wot   = (unsigned short*)take((size_t)EMB * EMB * 2);
    unsigned short* w1t   = (unsigned short*)take((size_t)EMB * FFD * 2);
    unsigned short* w2t   = (unsigned short*)take((size_t)FFD * EMB * 2);
    unsigned short* qbf   = (unsigned short*)take((size_t)HEADS * ROWS * HD * 2);
    unsigned short* kbf   = (unsigned short*)take((size_t)HEADS * ROWS * HD * 2);
    unsigned short* vbf   = (unsigned short*)take((size_t)HEADS * ROWS * HD * 2);
    unsigned short* attnb = (unsigned short*)take((size_t)ROWS * EMB * 2);
    float*          x2    = (float*)take((size_t)ROWS * EMB * 4);
    unsigned short* h2b   = (unsigned short*)take((size_t)ROWS * EMB * 2);
    unsigned short* y1b   = (unsigned short*)take((size_t)ROWS * FFD * 2);

    // 1) LN1 -> bf16
    ln_bf16_kernel<<<ROWS, 256, 0, stream>>>(x, g1, be1, hbf);

    // 2) weights -> transposed bf16 [N,K] (per head for QKV)
    const dim3 tb(32, 8);
    cvt_t_bf16_kernel<<<dim3(EMB / 32, HD / 32, HEADS), tb, 0, stream>>>(
        Wq, wqt, EMB, HD);
    cvt_t_bf16_kernel<<<dim3(EMB / 32, HD / 32, HEADS), tb, 0, stream>>>(
        Wk, wkt, EMB, HD);
    cvt_t_bf16_kernel<<<dim3(EMB / 32, HD / 32, HEADS), tb, 0, stream>>>(
        Wv, wvt, EMB, HD);
    cvt_t_bf16_kernel<<<dim3(EMB / 32, EMB / 32, 1), tb, 0, stream>>>(
        Wo, wot, EMB, EMB);
    cvt_t_bf16_kernel<<<dim3(EMB / 32, FFD / 32, 1), tb, 0, stream>>>(
        W1, w1t, EMB, FFD);
    cvt_t_bf16_kernel<<<dim3(FFD / 32, EMB / 32, 1), tb, 0, stream>>>(
        W2, w2t, FFD, EMB);

    // 3) Q/K/V per-head batched GEMMs: [R,1024] x [1024,64] -> [H,B*T,64] bf16
    const dim3 gq(HD / 64, ROWS / 128, HEADS);
    gemm_wmma_kernel<true, false, false><<<gq, 128, 0, stream>>>(
        hbf, wqt, nullptr, nullptr, qbf, ROWS, HD, EMB,
        (long)EMB * HD, (long)ROWS * HD);
    gemm_wmma_kernel<true, false, false><<<gq, 128, 0, stream>>>(
        hbf, wkt, nullptr, nullptr, kbf, ROWS, HD, EMB,
        (long)EMB * HD, (long)ROWS * HD);
    gemm_wmma_kernel<true, false, false><<<gq, 128, 0, stream>>>(
        hbf, wvt, nullptr, nullptr, vbf, ROWS, HD, EMB,
        (long)EMB * HD, (long)ROWS * HD);

    // 4) flash attention
    attn_wmma_kernel<<<dim3(TSEQ / 64, BATCH * HEADS), 128, 0, stream>>>(
        qbf, kbf, vbf, attnb);

    // 5) out-proj + residual: x2 = x + attn @ Wo + bo
    gemm_wmma_kernel<false, false, true>
        <<<dim3(EMB / 64, ROWS / 128, 1), 128, 0, stream>>>(
            attnb, wot, bo, x, x2, ROWS, EMB, EMB, 0, 0);

    // 6) LN2 -> bf16
    ln_bf16_kernel<<<ROWS, 256, 0, stream>>>(x2, g2, be2, h2b);

    // 7) FFN1: y1 = relu(h2 @ W1 + b1) -> bf16
    gemm_wmma_kernel<true, true, false>
        <<<dim3(FFD / 64, ROWS / 128, 1), 128, 0, stream>>>(
            h2b, w1t, b1, nullptr, y1b, ROWS, FFD, EMB, 0, 0);

    // 8) FFN2 + residual: out = x2 + y1 @ W2 + b2 (f32)
    gemm_wmma_kernel<false, false, true>
        <<<dim3(EMB / 64, ROWS / 128, 1), 128, 0, stream>>>(
            y1b, w2t, b2, x2, (float*)d_out, ROWS, EMB, FFD, 0, 0);
}
